// RNN_2095944040489
// MI455X (gfx1250) — matrix operand order repaired and checked by reading, not compiled, not run on hardware
//
#include <hip/hip_runtime.h>
#include <hip/hip_bf16.h>

// Problem constants (from reference)
#define Bsz   64
#define Tsz   1000
#define Hsz   512
#define Isz   32
#define Osz   32
#define ALPHA_C     0.05f
#define NOISE_STD_C 0.05f

typedef __attribute__((ext_vector_type(16))) __bf16 v16bf;
typedef __attribute__((ext_vector_type(8)))  float  v8f;

__device__ __forceinline__ unsigned short f2bf(float f) {
    unsigned int u = __float_as_uint(f);
    unsigned int r = (u + 0x7FFFu + ((u >> 16) & 1u)) >> 16;  // round-to-nearest-even
    return (unsigned short)r;
}

// Inverse of the 16-bit A-matrix VGPR layout (ISA 7.12.2):
// element (m, kk) of a 16x32 A tile -> (lane L, half-slot j)
__device__ __forceinline__ void amap(int m, int kk, int &L, int &j) {
    if (kk < 8)       { L = m;      j = kk;      }
    else if (kk < 16) { L = m + 16; j = kk - 8;  }
    else if (kk < 24) { L = m;      j = kk - 8;  }
    else              { L = m + 16; j = kk - 16; }
}

// ---------------------------------------------------------------------------
// Prep kernel: pack weights into WMMA B-tile lane-major layout (bf16) in d_ws.
//   Region 1: 32 N-tiles x 17 K-tiles (kt 0..15 = g-scaled wrec, kt 16 = wi)
//   Region 2: wout as 2 N-tiles x 16 K-tiles
// B-tile element mapping (32x16, K x N): lane L, slot j -> K = 8*(L/16)+j for j<8,
// K = 16+8*(L/16)+(j-8) for j>=8, N = L%16  (matches the A-tile layout of amap)
// ---------------------------------------------------------------------------
__global__ void prep_weights(const float* __restrict__ wi,
                             const float* __restrict__ wrec,
                             const float* __restrict__ wout,
                             const float* __restrict__ g,
                             unsigned short* __restrict__ ws) {
    const int NREC = 32 * 17 * 512;
    const int NOUT = 2 * 16 * 512;
    int idx = blockIdx.x * blockDim.x + threadIdx.x;
    if (idx < NREC) {
        int j  = idx & 15;
        int L  = (idx >> 4) & 31;
        int kt = (idx >> 9) % 17;
        int nt = idx / (17 * 512);
        int k  = (j < 8) ? ((L >> 4) << 3) + j : 16 + ((L >> 4) << 3) + (j - 8);  // 0..31
        int n  = nt * 16 + (L & 15);           // 0..511
        float v;
        if (kt < 16) {
            int K = kt * 32 + k;               // 0..511
            v = wrec[K * Hsz + n] * g[n];      // fold g into wrec columns
        } else {
            v = wi[k * Hsz + n];               // input GEMM as K-tile #16
        }
        ws[idx] = f2bf(v);
    } else if (idx < NREC + NOUT) {
        int i2 = idx - NREC;
        int j  = i2 & 15;
        int L  = (i2 >> 4) & 31;
        int kt = (i2 >> 9) & 15;
        int nt = i2 / (16 * 512);
        int k  = (j < 8) ? ((L >> 4) << 3) + j : 16 + ((L >> 4) << 3) + (j - 8);
        int K  = kt * 32 + k;                  // 0..511
        int n  = nt * 16 + (L & 15);           // 0..31
        ws[idx] = f2bf(wout[K * Osz + n]);
    }
}

// ---------------------------------------------------------------------------
// Main scan kernel: 4 blocks x 256 threads (8 wave32's). Block handles 16
// batch rows across all T steps. h state lives in registers (WMMA C layout).
// ---------------------------------------------------------------------------
__global__ __launch_bounds__(256) void rnn_scan(
    const float* __restrict__ x, const float* __restrict__ noise,
    const float* __restrict__ bvec, const float* __restrict__ tM,
    const float* __restrict__ h0,
    const unsigned short* __restrict__ wsB,   // wrec(+wi) B tiles, bf16
    const unsigned short* __restrict__ wsO,   // wout B tiles, bf16
    float* __restrict__ out) {

    // kt 0..15: r tiles (16x32 each); kt 16: x tile. Lane-major A layout.
    __shared__ __align__(32) unsigned short aPack[17 * 32 * 16];
    // wout B tiles staged once (loop-invariant): 2 nt x 16 kt x 512 halves.
    __shared__ __align__(32) unsigned short oPack[2 * 16 * 512];

    const int tid   = threadIdx.x;
    const int w     = tid >> 5;     // wave id 0..7 (wave32)
    const int lane  = tid & 31;
    const int hi    = lane >> 4;
    const int ln    = lane & 15;
    const int bbase = blockIdx.x * 16;

    // Per-lane column constants and register-resident h (WMMA C layout:
    // lane L vgpr v holds (M = v + 8*(L>=16), N = jcol)).
    float c1[4], c2[4], cb[4], hreg[4][8];
    int   jcol[4];
    #pragma unroll
    for (int tt = 0; tt < 4; ++tt) {
        int j = w * 64 + tt * 16 + ln;
        jcol[tt] = j;
        float a = ALPHA_C * (1.0f / tM[j]);
        c2[tt] = a;
        c1[tt] = 1.0f - a;
        cb[tt] = a * bvec[j];
        float h0v = h0[j];
        #pragma unroll
        for (int v = 0; v < 8; ++v) hreg[tt][v] = h0v;
    }

    // Stage wout B tiles into LDS (read every step by waves 0..1).
    for (int e = tid; e < 2 * 16 * 512; e += 256) oPack[e] = wsO[e];

    // Init aPack with r0 = relu(h0) broadcast over the 16 batch rows.
    for (int e = tid; e < 16 * Hsz; e += 256) {
        int m = e >> 9, k = e & 511;
        float r = h0[k]; r = r > 0.f ? r : 0.f;
        int L, j; amap(m, k & 31, L, j);
        aPack[((k >> 5) * 32 + L) * 16 + j] = f2bf(r);
    }
    // Init x tile (kt = 16) with x[:, 0, :].
    for (int e = tid; e < 16 * Isz; e += 256) {
        int m = e >> 5, kk = e & 31;
        float v = x[((bbase + m) * Tsz + 0) * Isz + kk];
        int L, j; amap(m, kk, L, j);
        aPack[(16 * 32 + L) * 16 + j] = f2bf(v);
    }
    __syncthreads();

    // Output row 0: relu(h0) @ wout  (waves 0..1; branch is wave-uniform so
    // EXEC stays all-ones inside WMMA as required).
    if (w < 2) {
        v8f acc = {0.f,0.f,0.f,0.f,0.f,0.f,0.f,0.f};
        #pragma unroll
        for (int kt = 0; kt < 16; ++kt) {
            v16bf A  = *(const v16bf*)(&aPack[(kt * 32 + lane) * 16]);
            v16bf Bt = *(const v16bf*)(&oPack[((w * 16 + kt) * 32 + lane) * 16]);
            acc = __builtin_amdgcn_wmma_f32_16x16x32_bf16(false, A, false, Bt,
                                                          (short)0, acc, false, false);
        }
        #pragma unroll
        for (int v = 0; v < 8; ++v) {
            int m = v + 8 * hi;
            out[((bbase + m) * Tsz + 0) * Osz + w * 16 + ln] = acc[v];
        }
    }

    for (int s = 0; s < Tsz - 1; ++s) {
        // ---- Prefetch: issue HBM-cold noise[:,s] and x[:,s+1] loads NOW so
        // their latency hides under the 68-WMMA phase-1 GEMM. ----
        float nreg[4][8];
        #pragma unroll
        for (int tt = 0; tt < 4; ++tt) {
            int j = jcol[tt];
            #pragma unroll
            for (int v = 0; v < 8; ++v) {
                int m = v + 8 * hi;
                nreg[tt][v] = noise[((bbase + m) * Tsz + s) * Hsz + j];
            }
        }
        float xreg[2];
        #pragma unroll
        for (int c = 0; c < 2; ++c) {
            int e = tid + c * 256;                 // e < 512
            int m = e >> 5, kk = e & 31;
            xreg[c] = x[((bbase + m) * Tsz + (s + 1)) * Isz + kk];
        }

        // ---- Phase 1: acc = r @ (g.*wrec) + x @ wi  (17 fused K-tiles) ----
        // Fully unrolled: exposes all 136 B-fragment global loads to the
        // scheduler so L2 latency is hidden behind the WMMA stream.
        v8f acc[4];
        #pragma unroll
        for (int tt = 0; tt < 4; ++tt)
            acc[tt] = (v8f){0.f,0.f,0.f,0.f,0.f,0.f,0.f,0.f};
        #pragma unroll
        for (int kt = 0; kt <= 16; ++kt) {
            v16bf A = *(const v16bf*)(&aPack[(kt * 32 + lane) * 16]);
            #pragma unroll
            for (int tt = 0; tt < 4; ++tt) {
                int nt = w * 4 + tt;
                v16bf Bt = *(const v16bf*)(&wsB[((nt * 17 + kt) * 32 + lane) * 16]);
                acc[tt] = __builtin_amdgcn_wmma_f32_16x16x32_bf16(false, A, false, Bt,
                                                                  (short)0, acc[tt],
                                                                  false, false);
            }
        }
        __syncthreads();   // all aPack reads complete before rewriting it

        // ---- Phase 2: in-register h update, repack r into aPack ----
        #pragma unroll
        for (int tt = 0; tt < 4; ++tt) {
            int j  = jcol[tt];
            int kt = j >> 5, kk = j & 31;
            #pragma unroll
            for (int v = 0; v < 8; ++v) {
                int m = v + 8 * hi;
                float hn = c1[tt] * hreg[tt][v] + c2[tt] * acc[tt][v]
                         + cb[tt] + NOISE_STD_C * nreg[tt][v];
                hreg[tt][v] = hn;
                float r = hn > 0.f ? hn : 0.f;
                int L, jj; amap(m, kk, L, jj);
                aPack[(kt * 32 + L) * 16 + jj] = f2bf(r);
            }
        }
        // store prefetched x[:, s+1, :] into kt=16
        #pragma unroll
        for (int c = 0; c < 2; ++c) {
            int e = tid + c * 256;
            int m = e >> 5, kk = e & 31;
            int L, j; amap(m, kk, L, j);
            aPack[(16 * 32 + L) * 16 + j] = f2bf(xreg[c]);
        }
        __syncthreads();   // new r/x tiles visible

        // ---- Phase 3: out[:, s+1, :] = relu(h_new) @ wout (waves 0..1) ----
        if (w < 2) {
            v8f oacc = {0.f,0.f,0.f,0.f,0.f,0.f,0.f,0.f};
            #pragma unroll
            for (int kt = 0; kt < 16; ++kt) {
                v16bf A  = *(const v16bf*)(&aPack[(kt * 32 + lane) * 16]);
                v16bf Bt = *(const v16bf*)(&oPack[((w * 16 + kt) * 32 + lane) * 16]);
                oacc = __builtin_amdgcn_wmma_f32_16x16x32_bf16(false, A, false, Bt,
                                                               (short)0, oacc,
                                                               false, false);
            }
            #pragma unroll
            for (int v = 0; v < 8; ++v) {
                int m = v + 8 * hi;
                out[((bbase + m) * Tsz + (s + 1)) * Osz + w * 16 + ln] = oacc[v];
            }
        }
    }
}

extern "C" void kernel_launch(void* const* d_in, const int* in_sizes, int n_in,
                              void* d_out, int out_size, void* d_ws, size_t ws_size,
                              hipStream_t stream) {
    const float* x     = (const float*)d_in[0];
    const float* noise = (const float*)d_in[1];
    const float* wi    = (const float*)d_in[2];
    const float* wrec  = (const float*)d_in[3];
    const float* wout  = (const float*)d_in[4];
    const float* b     = (const float*)d_in[5];
    const float* g     = (const float*)d_in[6];
    const float* tM    = (const float*)d_in[7];
    const float* h0    = (const float*)d_in[8];
    (void)in_sizes; (void)n_in; (void)out_size; (void)ws_size;

    unsigned short* wsB = (unsigned short*)d_ws;          // 32*17*512 bf16
    unsigned short* wsO = wsB + 32 * 17 * 512;            // 2*16*512 bf16
    float* out = (float*)d_out;

    const int NTOT = 32 * 17 * 512 + 2 * 16 * 512;
    prep_weights<<<(NTOT + 255) / 256, 256, 0, stream>>>(wi, wrec, wout, g, wsB);
    rnn_scan<<<4, 256, 0, stream>>>(x, noise, b, tM, h0, wsB, wsO, out);
}
